// SSP_57664230916484
// MI455X (gfx1250) — matrix operand order repaired and checked by reading, not compiled, not run on hardware
//
#include <hip/hip_runtime.h>
#include <hip/hip_bf16.h>

// ---------------------------------------------------------------------------
// 2-layer GCN (GraphConv norm='both') for MI455X / gfx1250.
// GEMMs use v_wmma_f32_16x16x32_bf16 (f32 accumulate):
//   - A strip (16 x K) converted to bf16 in LDS once per block, fragments read
//     back as ds_load_b128.
//   - B (weights) pre-packed into exact WMMA fragment order -> each wave loads
//     its fragment as two coalesced global_load_b128.
// SpMM is edge-parallel f32 atomic scatter-add (accumulators are L2-resident:
// 192MB L2 holds every intermediate, so gathers/atomics rarely touch HBM).
// ---------------------------------------------------------------------------

typedef __attribute__((ext_vector_type(16))) __bf16 v16bf;
typedef __attribute__((ext_vector_type(8)))  float  v8f;

struct U128 { unsigned int w[4]; };

union Frag16 {            // naturally 32-byte aligned via v16bf member
  v16bf v;
  unsigned short u[16];
  U128 q[2];
};

__device__ __forceinline__ unsigned short f2bf(float f) {
  // round-to-nearest-even truncation f32 -> bf16
  unsigned int x = __float_as_uint(f);
  x += 0x7FFFu + ((x >> 16) & 1u);
  return (unsigned short)(x >> 16);
}

// ------------------------- degree / norm kernels ---------------------------

__global__ void degree_k(const int* __restrict__ src, const int* __restrict__ dst,
                         float* __restrict__ deg_out, float* __restrict__ deg_in, int E) {
  int e = blockIdx.x * blockDim.x + threadIdx.x;
  if (e >= E) return;
  atomicAdd(deg_out + src[e], 1.0f);
  atomicAdd(deg_in + dst[e], 1.0f);
}

__global__ void invsqrt_k(float* __restrict__ d, int n) {
  int i = blockIdx.x * blockDim.x + threadIdx.x;
  if (i >= n) return;
  float v = d[i];
  d[i] = rsqrtf(v < 1.0f ? 1.0f : v);   // clip(deg, 1)^-1/2
}

// ----------------- weight pre-pack: f32 [K x Nin] -> bf16 fragments --------
// Output layout: Bp[(((ntile * KS) + ks) * 32 + lane) * 16 + i]
// where fragment element (lane, i) holds B[krow, col]:
//   col  = ntile*16 + (lane & 15)
//   krow = ks*32 + (lane >= 16 ? 8 : 0) + (i & 7) + (i >= 8 ? 16 : 0)   (ISA 7.12.2, 16-bit B 32x16)
// Zero-padded for col >= Nin, so the GEMM needs no bounds checks.

__global__ void pack_b_k(const float* __restrict__ W, unsigned short* __restrict__ Bp,
                         int K, int Nin, int Ntiles) {
  int t = blockIdx.x * blockDim.x + threadIdx.x;
  const int KS = K >> 5;
  const int total = Ntiles * KS * 32 * 16;
  if (t >= total) return;
  const int i = t & 15;
  const int lane = (t >> 4) & 31;
  const int frag = t >> 9;              // ntile * KS + ks
  const int ks = frag % KS;
  const int ntile = frag / KS;
  const int col = ntile * 16 + (lane & 15);
  const int krow = ks * 32 + ((lane >> 4) << 3) + (i & 7) + ((i >> 3) << 4);
  const float v = (col < Nin) ? W[krow * Nin + col] : 0.0f;
  Bp[t] = f2bf(v);
}

// ------------------------------ WMMA GEMM ----------------------------------
// C[m, col] = rowscale[m] * sum_k A[m,k] * B[k,col]
// Block: (32, NTILES) waves; one 16-row strip per block, one 16-col tile per wave.

template <int K, int NTILES, int NOUT>
__global__ __launch_bounds__(256) void gemm_wmma_bf16(
    const float* __restrict__ A, const unsigned short* __restrict__ Bp,
    const float* __restrict__ rowscale, float* __restrict__ C, int M) {
  __shared__ __align__(16) unsigned short aTile[16 * K];

  const int mbase = blockIdx.x * 16;
  const int tid = threadIdx.y * 32 + threadIdx.x;
  const int nthr = blockDim.y * 32;
  for (int i = tid; i < 16 * K; i += nthr) {
    int row = i / K;
    int k = i - row * K;
    int m = mbase + row;
    if (m >= M) m = M - 1;                 // safe clamp (M is multiple of 16 here)
    aTile[i] = f2bf(A[(size_t)m * K + k]);
  }
  __syncthreads();

  const int lane = threadIdx.x;
  const int half = lane >> 4;              // 0: lanes 0-15, 1: lanes 16-31
  const int l15 = lane & 15;
  const int col = threadIdx.y * 16 + l15;

  // Base of this wave's packed-B fragments (16 bf16 per lane, contiguous).
  const unsigned short* bbase =
      Bp + ((size_t)threadIdx.y * (K / 32) * 32 + lane) * 16;

  v8f acc = {};
#pragma unroll
  for (int ks = 0; ks < K / 32; ++ks) {
    Frag16 a, b;
    // A fragment (ISA 7.12.2, 16-bit A 16x32): lane half selects K sub-range.
#pragma unroll
    for (int g = 0; g < 2; ++g) {
      const int kb = ks * 32 + half * 8 + g * 16;
      a.q[g] = *(const U128*)&aTile[l15 * K + kb];
    }
    // B fragment: pre-packed, two b128 loads per lane, coalesced across wave.
    const unsigned short* bp = bbase + (size_t)ks * 32 * 16;
    b.q[0] = *(const U128*)bp;
    b.q[1] = *(const U128*)(bp + 8);
    acc = __builtin_amdgcn_wmma_f32_16x16x32_bf16(false, a.v, false, b.v,
                                                  (short)0, acc, false, false);
  }

  if (col < NOUT) {
#pragma unroll
    for (int v = 0; v < 8; ++v) {
      const int m = mbase + half * 8 + v;  // C layout: VGPR v -> M = v + 8*half
      if (m < M)
        C[(size_t)m * NOUT + col] = acc[v] * rowscale[m];
    }
  }
}

// ------------------------------ SpMM scatter -------------------------------
// One thread per (edge, float4 chunk): coalesced row gather, f32 atomics out.

template <int F>
__global__ void spmm_k(const float* __restrict__ X, const int* __restrict__ src,
                       const int* __restrict__ dst, float* __restrict__ Y,
                       unsigned int total) {
  unsigned int t = blockIdx.x * blockDim.x + threadIdx.x;
  if (t >= total) return;
  constexpr unsigned int C = F / 4;
  unsigned int e = t / C;
  unsigned int c = t - e * C;
  const int s = src[e], d = dst[e];
  const float4 vv = *((const float4*)(X + (size_t)s * F) + c);
  float* y = Y + (size_t)d * F + c * 4;
  atomicAdd(y + 0, vv.x);
  atomicAdd(y + 1, vv.y);
  atomicAdd(y + 2, vv.z);
  atomicAdd(y + 3, vv.w);
}

// --------------------- epilogue 1: dst-norm + bias + relu ------------------

__global__ void post_relu_k(float* __restrict__ X, const float* __restrict__ rs_in,
                            const float* __restrict__ b, int total) {
  int t = blockIdx.x * blockDim.x + threadIdx.x;
  if (t >= total) return;
  int node = t >> 7;                       // F = 128
  int col = t & 127;
  float v = X[t] * rs_in[node] + b[col];
  X[t] = v > 0.0f ? v : 0.0f;
}

// --------------- epilogue 2: dst-norm + bias + log_softmax -----------------

__global__ void logsoftmax_k(const float* __restrict__ agg, const float* __restrict__ rs_in,
                             const float* __restrict__ b, float* __restrict__ out, int N) {
  int node = blockIdx.x * blockDim.x + threadIdx.x;
  if (node >= N) return;
  const float r = rs_in[node];
  float v[40];
  float mx = -3.402823466e38f;
#pragma unroll
  for (int c = 0; c < 40; ++c) {
    v[c] = agg[(size_t)node * 40 + c] * r + b[c];
    mx = fmaxf(mx, v[c]);
  }
  float s = 0.0f;
#pragma unroll
  for (int c = 0; c < 40; ++c) s += __expf(v[c] - mx);
  const float lse = mx + __logf(s);
#pragma unroll
  for (int c = 0; c < 40; ++c) out[(size_t)node * 40 + c] = v[c] - lse;
}

// ------------------------------- launcher ----------------------------------

extern "C" void kernel_launch(void* const* d_in, const int* in_sizes, int n_in,
                              void* d_out, int out_size, void* d_ws, size_t ws_size,
                              hipStream_t stream) {
  const float* h  = (const float*)d_in[0];   // [N,256]
  const float* W1 = (const float*)d_in[1];   // [256,128]
  const float* b1 = (const float*)d_in[2];   // [128]
  const float* W2 = (const float*)d_in[3];   // [128,40]
  const float* b2 = (const float*)d_in[4];   // [40]
  const int* src  = (const int*)d_in[5];     // [E]
  const int* dst  = (const int*)d_in[6];     // [E]
  float* out = (float*)d_out;

  const int IN = 256, HID = 128, OUT = 40;
  const int NT1 = 8, NT2 = 3;                // N-tiles per layer (128/16, ceil(40/16))
  const int N = in_sizes[0] / IN;            // 100000 (multiple of 16)
  const int E = in_sizes[5];                 // 3200000

  // Workspace layout (~108 MB total):
  //   [0, 0.8MB)   rs_out | rs_in   (2N f32, contiguous)
  //   [1MB)        W1 packed bf16 fragments (64KB); [1.25MB) W2 packed (12KB)
  //   [2MB)        h1  (N*128 f32 = 51.2MB)  -- dead after SpMM1, then reused:
  //                  x2   at [2MB)  (N*40 f32 = 16MB)
  //                  agg2 at [22MB) (N*40 f32 = 16MB)
  //   [56MB)       x1 / agg1 (N*128 f32 = 51.2MB)
  char* ws = (char*)d_ws;
  float* rs      = (float*)ws;
  float* rs_out  = rs;
  float* rs_in   = rs + N;
  unsigned short* W1p = (unsigned short*)(ws + (1ll << 20));
  unsigned short* W2p = (unsigned short*)(ws + (1ll << 20) + (256 * 1024));
  float* h1   = (float*)(ws + (2ll  << 20));
  float* x1   = (float*)(ws + (56ll << 20));
  float* x2   = (float*)(ws + (2ll  << 20));
  float* agg2 = (float*)(ws + (22ll << 20));

  const int bs = 256;

  // degrees -> rsqrt norms
  (void)hipMemsetAsync(rs, 0, sizeof(float) * 2 * (size_t)N, stream);
  degree_k<<<(E + bs - 1) / bs, bs, 0, stream>>>(src, dst, rs_out, rs_in, E);
  invsqrt_k<<<(2 * N + bs - 1) / bs, bs, 0, stream>>>(rs, 2 * N);

  // weights -> bf16 WMMA fragments (zero-padded in N)
  {
    int tot1 = NT1 * (IN / 32) * 32 * 16;    // 32768
    int tot2 = NT2 * (HID / 32) * 32 * 16;   // 6144
    pack_b_k<<<(tot1 + bs - 1) / bs, bs, 0, stream>>>(W1, W1p, IN, HID, NT1);
    pack_b_k<<<(tot2 + bs - 1) / bs, bs, 0, stream>>>(W2, W2p, HID, OUT, NT2);
  }

  // layer 1: h1 = rs_out .* (h @ W1)   [WMMA]
  gemm_wmma_bf16<256, 8, 128><<<N / 16, dim3(32, NT1), 0, stream>>>(h, W1p, rs_out, h1, N);

  // SpMM1: x1[dst] += h1[src]
  (void)hipMemsetAsync(x1, 0, sizeof(float) * (size_t)N * HID, stream);
  {
    unsigned int total = (unsigned int)E * (HID / 4);
    spmm_k<128><<<(total + bs - 1) / bs, bs, 0, stream>>>(h1, src, dst, x1, total);
  }
  post_relu_k<<<(N * HID + bs - 1) / bs, bs, 0, stream>>>(x1, rs_in, b1, N * HID);

  // layer 2: x2 = rs_out .* (x1 @ W2)  [WMMA]
  gemm_wmma_bf16<128, 3, 40><<<N / 16, dim3(32, NT2), 0, stream>>>(x1, W2p, rs_out, x2, N);

  // SpMM2: agg2[dst] += x2[src]
  (void)hipMemsetAsync(agg2, 0, sizeof(float) * (size_t)N * OUT, stream);
  {
    unsigned int total = (unsigned int)E * (OUT / 4);
    spmm_k<40><<<(total + bs - 1) / bs, bs, 0, stream>>>(x2, src, dst, agg2, total);
  }

  // dst norm + bias + log_softmax -> out
  logsoftmax_k<<<(N + bs - 1) / bs, bs, 0, stream>>>(agg2, rs_in, b2, out, N);
}
